// Generator_22411139350698
// MI455X (gfx1250) — matrix operand order repaired and checked by reading, not compiled, not run on hardware
//
#include <hip/hip_runtime.h>
#include <hip/hip_bf16.h>
#include <math.h>

// ---------------------------------------------------------------------------
// Types for CDNA5 WMMA (gfx1250, wave32)
// ---------------------------------------------------------------------------
typedef __attribute__((ext_vector_type(16))) __bf16 v16bf;
typedef __attribute__((ext_vector_type(8)))  __bf16 v8bf;
typedef __attribute__((ext_vector_type(8)))  float  v8f;

#define BATCH 2048
#define TSTEP 64
#define HIDD  128
#define MLPW  512
#define NOISE 16

// Tensor Data Mover availability (ROCm7.2 clang-22: 5-arg; clang-23: 6-arg)
#if defined(__HIP_DEVICE_COMPILE__) && __has_builtin(__builtin_amdgcn_tensor_load_to_lds) && \
    __has_builtin(__builtin_amdgcn_s_wait_tensorcnt)
#define USE_TDM 1
#else
#define USE_TDM 0
#endif

#if USE_TDM
typedef __attribute__((ext_vector_type(4))) unsigned int tdm_v4u;
typedef __attribute__((ext_vector_type(8))) int          tdm_v8i;
typedef __attribute__((ext_vector_type(4))) int          tdm_v4i;

// Issue one 2D TDM tile load: rows x 16 dwords (rows x 32 bf16), row stride
// k_elems bf16 (= k_elems/2 dwords), global -> LDS, contiguous in LDS.
// D# layout per CDNA5 ISA ch.8: group0 = {count|flags, lds_addr, gaddr_lo,
// gaddr_hi|type<<30}; group1 packs data_size/dims/tile/stride.
__device__ __forceinline__ void tdm_load_tile(unsigned ldsOff, const __bf16* gsrc,
                                              unsigned rows, unsigned k_elems) {
  unsigned long long ga = (unsigned long long)(const void*)gsrc;
  unsigned stride_u32 = k_elems >> 1;
  tdm_v4u g0;
  g0[0] = 1u;                                    // count=1, no gather, user mode
  g0[1] = ldsOff;                                // lds_addr (bytes)
  g0[2] = (unsigned)ga;                          // global_addr[31:0]
  g0[3] = (unsigned)(ga >> 32) | 0x80000000u;    // global_addr[56:32] | type=2
  tdm_v8i g1;
  g1[0] = 0x00020000;                            // data_size=4B, mask=0, no pad
  g1[1] = (int)(16u << 16);                      // tensor_dim0 = 16 dwords
  g1[2] = (int)(rows << 16);                     // tensor_dim1 = rows
  g1[3] = (int)(16u << 16);                      // tile_dim0 = 16 dwords
  g1[4] = (int)rows;                             // tile_dim1 = rows, tile_dim2=0
  g1[5] = (int)stride_u32;                       // tensor_dim0_stride (dwords)
  g1[6] = 0;
  g1[7] = 0;
  tdm_v4i z4 = {0, 0, 0, 0};
#if __clang_major__ >= 23
  tdm_v8i z8 = {0, 0, 0, 0, 0, 0, 0, 0};
  __builtin_amdgcn_tensor_load_to_lds(g0, g1, z4, z4, z8, 0);
#else
  __builtin_amdgcn_tensor_load_to_lds(g0, g1, z4, z4, 0);
#endif
}
#endif  // USE_TDM

// ---------------------------------------------------------------------------
// Device helpers
// ---------------------------------------------------------------------------
__device__ __forceinline__ float lipswish_f(float x) {
  return 0.909f * x / (1.0f + __expf(-x));
}

// A-fragment (16x32 bf16, M x K) per ISA 7.12.2:
//   lanes 0-15 : row M=lane, K=0..7 in v[0..7],  K=16..23 in v[8..15]
//   lanes 16-31: row M=lane-16, K=8..15 / K=24..31
__device__ __forceinline__ v16bf load_a_frag(const __bf16* As, int rowBase, int lane) {
  const int r   = lane & 15;
  const int hlf = lane >> 4;
  const __bf16* p = As + (rowBase + r) * 32 + hlf * 8;
  v16bf f;
  *((v8bf*)&f)     = *(const v8bf*)(p);        // K = k0 .. k0+7
  *(((v8bf*)&f)+1) = *(const v8bf*)(p + 16);   // K = k0+16 .. k0+23
  return f;
}

// B-fragment (32x16 bf16, K x N): lanes 0-15 col N=lane K=0..7 / K=16..23;
// lanes 16-31 col N=lane-16 K=8..15 / K=24..31. Bs tile is W^T rows x 32 bf16.
__device__ __forceinline__ v16bf load_b_frag(const __bf16* Bs, int colBase, int lane) {
  const int n   = lane & 15;
  const int hlf = lane >> 4;
  const __bf16* p = Bs + (colBase + n) * 32 + hlf * 8;
  v16bf f;
  *((v8bf*)&f)     = *(const v8bf*)(p);
  *(((v8bf*)&f)+1) = *(const v8bf*)(p + 16);
  return f;
}

// ---------------------------------------------------------------------------
// WMMA GEMM:  C(M x N) = act( A(M x K, bf16) * W(K x N) + bias )
//   Block: 256 threads = 8 waves (2 x 4), block tile 64 x 128,
//   each wave computes 32 x 32 via 2x2 v_wmma_f32_16x16x32_bf16.
//   Tiles staged via TDM (double-buffered, prefetch next chunk) when
//   available; fallback is vector-load + ds_store staging.
//   ACT: 0 = none, 1 = lipswish, 2 = tanh.
// ---------------------------------------------------------------------------
template<int ACT, bool WF, bool WB>
__global__ __launch_bounds__(256) void gemm_wmma_kernel(
    const __bf16* __restrict__ A, const __bf16* __restrict__ Wt,
    const float* __restrict__ bias,
    float* __restrict__ outF, __bf16* __restrict__ outB,
    int K, int N)
{
  __shared__ alignas(16) __bf16 As[2][64 * 32];
  __shared__ alignas(16) __bf16 Bs[2][128 * 32];

  const int tid     = threadIdx.x;
  const int lane    = tid & 31;
  const int wave    = tid >> 5;
  const int waveRow = wave >> 2;   // 0..1
  const int waveCol = wave & 3;    // 0..3
  const int rowBlk  = blockIdx.y * 64;
  const int colBlk  = blockIdx.x * 128;
  const int nk      = K >> 5;

  const __bf16* Abase = A  + (size_t)rowBlk * K;
  const __bf16* Bbase = Wt + (size_t)colBlk * K;

  v8f acc[2][2] = {};

#if USE_TDM
  const bool leader = (__builtin_amdgcn_readfirstlane(tid) < 32);
  const unsigned asOff0 = (unsigned)(unsigned long long)(const void*)&As[0][0];
  const unsigned asOff1 = (unsigned)(unsigned long long)(const void*)&As[1][0];
  const unsigned bsOff0 = (unsigned)(unsigned long long)(const void*)&Bs[0][0];
  const unsigned bsOff1 = (unsigned)(unsigned long long)(const void*)&Bs[1][0];
  if (leader) {
    tdm_load_tile(asOff0, Abase, 64, (unsigned)K);
    tdm_load_tile(bsOff0, Bbase, 128, (unsigned)K);
  }
#else
  const int ldr = tid >> 2;   // 0..63
  const int lsg = tid & 3;    // 16B segment within a 32-bf16 row
#endif

  for (int kc = 0; kc < nk; ++kc) {
    const int cur = kc & 1;
#if USE_TDM
    const int nxt = kc + 1;
    if (leader) {
      if (nxt < nk) {
        tdm_load_tile(nxt & 1 ? asOff1 : asOff0, Abase + (size_t)nxt * 32, 64, (unsigned)K);
        tdm_load_tile(nxt & 1 ? bsOff1 : bsOff0, Bbase + (size_t)nxt * 32, 128, (unsigned)K);
        __builtin_amdgcn_s_wait_tensorcnt(2);   // chunk kc's two DMAs done
      } else {
        __builtin_amdgcn_s_wait_tensorcnt(0);
      }
    }
    __syncthreads();                            // publish chunk kc to all waves
#else
    *(float4*)&As[cur][ldr * 32 + lsg * 8] =
        *(const float4*)&Abase[(size_t)ldr * K + kc * 32 + lsg * 8];
    *(float4*)&Bs[cur][ldr * 32 + lsg * 8] =
        *(const float4*)&Bbase[(size_t)ldr * K + kc * 32 + lsg * 8];
    *(float4*)&Bs[cur][(ldr + 64) * 32 + lsg * 8] =
        *(const float4*)&Bbase[(size_t)(ldr + 64) * K + kc * 32 + lsg * 8];
    __syncthreads();
#endif

    const __bf16* AsC = &As[cur][0];
    const __bf16* BsC = &Bs[cur][0];
    v16bf af[2], bf[2];
    af[0] = load_a_frag(AsC, waveRow * 32 + 0,  lane);
    af[1] = load_a_frag(AsC, waveRow * 32 + 16, lane);
    bf[0] = load_b_frag(BsC, waveCol * 32 + 0,  lane);
    bf[1] = load_b_frag(BsC, waveCol * 32 + 16, lane);

#pragma unroll
    for (int i = 0; i < 2; ++i)
#pragma unroll
      for (int j = 0; j < 2; ++j)
        acc[i][j] = __builtin_amdgcn_wmma_f32_16x16x32_bf16(
            false, af[i], false, bf[j], (short)0, acc[i][j], false, false);
    __syncthreads();                            // done reading buf[cur]
  }

  // Epilogue: C/D layout — VGPR r holds M=r (lanes 0-15) / M=r+8 (lanes 16-31),
  // N = lane % 16.
  const int hlf = lane >> 4;
  const int nn  = lane & 15;
#pragma unroll
  for (int i = 0; i < 2; ++i)
#pragma unroll
    for (int j = 0; j < 2; ++j) {
      const int gcol = colBlk + waveCol * 32 + j * 16 + nn;
      const float bv = bias[gcol];
#pragma unroll
      for (int r = 0; r < 8; ++r) {
        const int grow = rowBlk + waveRow * 32 + i * 16 + hlf * 8 + r;
        float v = acc[i][j][r] + bv;
        if (ACT == 1) v = lipswish_f(v);
        if (ACT == 2) v = tanhf(v);
        const size_t o = (size_t)grow * N + gcol;
        if (WB) outB[o] = (__bf16)v;
        if (WF) outF[o] = v;
      }
    }
}

// ---------------------------------------------------------------------------
// One-time preparation kernels
// ---------------------------------------------------------------------------
__global__ void conv_wt_kernel(const float* __restrict__ W, __bf16* __restrict__ Wt,
                               int K, int N, int rowOff) {
  int idx = blockIdx.x * blockDim.x + threadIdx.x;
  if (idx >= K * N) return;
  int k = idx / N, n = idx - k * N;
  Wt[(size_t)n * K + k] = (__bf16)W[(size_t)(k + rowOff) * N + n];
}

__global__ void conv_bf_kernel(const float* __restrict__ src, __bf16* __restrict__ dst, int n) {
  int i = blockIdx.x * blockDim.x + threadIdx.x;
  if (i < n) dst[i] = (__bf16)src[i];
}

// Fold t-column of W0 into bias: beff[t][n] = b0[n] + ts[t] * W0[0][n]
__global__ void bias_eff_kernel(const float* __restrict__ b0, const float* __restrict__ w0row,
                                const float* __restrict__ ts, float* __restrict__ out) {
  int idx = blockIdx.x * blockDim.x + threadIdx.x;   // 64*512
  if (idx >= TSTEP * MLPW) return;
  int t = idx >> 9, n = idx & 511;
  out[idx] = b0[n] + ts[t] * w0row[n];
}

// ---------------------------------------------------------------------------
// Sequential scan elementwise kernels
// ---------------------------------------------------------------------------
__global__ void init_state_kernel(const float* __restrict__ x0,
                                  float* __restrict__ yhat0, float* __restrict__ xs) {
  int idx = blockIdx.x * blockDim.x + threadIdx.x;   // B*HID
  if (idx >= BATCH * HIDD) return;
  int b = idx >> 7, h = idx & 127;
  float v = x0[idx];
  yhat0[idx] = v;
  xs[((size_t)b * TSTEP + 0) * HIDD + h] = v;
}

__global__ void step_yhat_kernel(const float* __restrict__ y, const float* __restrict__ yhat,
                                 const float* __restrict__ f, const float* __restrict__ g,
                                 const float* __restrict__ dWk, const float* __restrict__ ts, int k,
                                 float* __restrict__ gdw, float* __restrict__ yhat1,
                                 __bf16* __restrict__ yhat1_bf) {
  int idx = blockIdx.x * blockDim.x + threadIdx.x;   // B*HID
  if (idx >= BATCH * HIDD) return;
  int b = idx >> 7;
  float dt  = ts[k + 1] - ts[k];
  float sdt = sqrtf(dt);
  const float* gp = g + (size_t)idx * NOISE;
  const float* wp = dWk + (size_t)b * NOISE;
  float s = 0.f;
#pragma unroll
  for (int n = 0; n < NOISE; ++n) s += gp[n] * (wp[n] * sdt);
  gdw[idx] = s;
  float v = 2.f * y[idx] - yhat[idx] + f[idx] * dt + s;
  yhat1[idx]    = v;
  yhat1_bf[idx] = (__bf16)v;
}

__global__ void step_y_kernel(const float* __restrict__ y,
                              const float* __restrict__ f_old, const float* __restrict__ f_new,
                              const float* __restrict__ gdw_old, const float* __restrict__ g_new,
                              const float* __restrict__ dWk, const float* __restrict__ ts, int k,
                              float* __restrict__ y_new, float* __restrict__ xs) {
  int idx = blockIdx.x * blockDim.x + threadIdx.x;   // B*HID
  if (idx >= BATCH * HIDD) return;
  int b = idx >> 7, h = idx & 127;
  float dt  = ts[k + 1] - ts[k];
  float sdt = sqrtf(dt);
  const float* gp = g_new + (size_t)idx * NOISE;
  const float* wp = dWk + (size_t)b * NOISE;
  float s = 0.f;
#pragma unroll
  for (int n = 0; n < NOISE; ++n) s += gp[n] * (wp[n] * sdt);
  float v = y[idx] + 0.5f * (f_old[idx] + f_new[idx]) * dt + 0.5f * (gdw_old[idx] + s);
  y_new[idx] = v;
  xs[((size_t)b * TSTEP + (k + 1)) * HIDD + h] = v;
}

__global__ void readout_kernel(const float* __restrict__ xs, const float* __restrict__ ts,
                               const float* __restrict__ rW, const float* __restrict__ rb,
                               float* __restrict__ out) {
  int idx = blockIdx.x * blockDim.x + threadIdx.x;   // B*T
  if (idx >= BATCH * TSTEP) return;
  int t = idx & 63;
  const float* xr = xs + (size_t)idx * HIDD;
  float o[8];
#pragma unroll
  for (int d = 0; d < 8; ++d) o[d] = rb[d];
  for (int h = 0; h < HIDD; ++h) {
    float xv = xr[h];
#pragma unroll
    for (int d = 0; d < 8; ++d) o[d] = fmaf(xv, rW[h * 8 + d], o[d]);
  }
  float* op = out + (size_t)idx * 9;
  op[0] = ts[t];
#pragma unroll
  for (int d = 0; d < 8; ++d) op[1 + d] = o[d];
}

// ---------------------------------------------------------------------------
// Host driver
// ---------------------------------------------------------------------------
extern "C" void kernel_launch(void* const* d_in, const int* in_sizes, int n_in,
                              void* d_out, int out_size, void* d_ws, size_t ws_size,
                              hipStream_t stream) {
  (void)in_sizes; (void)n_in; (void)out_size; (void)ws_size;
  const float* ts    = (const float*)d_in[0];
  const float* noise = (const float*)d_in[2];
  const float* dW    = (const float*)d_in[3];
  const float* iW0 = (const float*)d_in[4],  *ib0 = (const float*)d_in[5];
  const float* iWh = (const float*)d_in[6],  *ibh = (const float*)d_in[7];
  const float* iWo = (const float*)d_in[8],  *ibo = (const float*)d_in[9];
  const float* fW0 = (const float*)d_in[10], *fb0 = (const float*)d_in[11];
  const float* fWh = (const float*)d_in[12], *fbh = (const float*)d_in[13];
  const float* fWo = (const float*)d_in[14], *fbo = (const float*)d_in[15];
  const float* gW0 = (const float*)d_in[16], *gb0 = (const float*)d_in[17];
  const float* gWh = (const float*)d_in[18], *gbh = (const float*)d_in[19];
  const float* gWo = (const float*)d_in[20], *gbo = (const float*)d_in[21];
  const float* rW  = (const float*)d_in[22], *rb  = (const float*)d_in[23];
  float* out = (float*)d_out;

  // ---- workspace layout (256B-aligned bump allocator) ----
  char* ws = (char*)d_ws;
  size_t off = 0;
  auto alloc = [&](size_t bytes) -> char* {
    off = (off + 255) & ~(size_t)255;
    char* p = ws + off;
    off += bytes;
    return p;
  };
  __bf16* iW0t  = (__bf16*)alloc((size_t)512 * 64  * 2);
  __bf16* iWh0t = (__bf16*)alloc((size_t)512 * 512 * 2);
  __bf16* iWh1t = (__bf16*)alloc((size_t)512 * 512 * 2);
  __bf16* iWot  = (__bf16*)alloc((size_t)128 * 512 * 2);
  __bf16* fW0t  = (__bf16*)alloc((size_t)512 * 128 * 2);
  __bf16* fWh0t = (__bf16*)alloc((size_t)512 * 512 * 2);
  __bf16* fWh1t = (__bf16*)alloc((size_t)512 * 512 * 2);
  __bf16* fWot  = (__bf16*)alloc((size_t)128 * 512 * 2);
  __bf16* gW0t  = (__bf16*)alloc((size_t)512 * 128 * 2);
  __bf16* gWh0t = (__bf16*)alloc((size_t)512 * 512 * 2);
  __bf16* gWh1t = (__bf16*)alloc((size_t)512 * 512 * 2);
  __bf16* gWot  = (__bf16*)alloc((size_t)2048 * 512 * 2);
  __bf16* actA  = (__bf16*)alloc((size_t)BATCH * MLPW * 2);
  __bf16* actB  = (__bf16*)alloc((size_t)BATCH * MLPW * 2);
  __bf16* x_bf  = (__bf16*)alloc((size_t)BATCH * HIDD * 2);
  __bf16* noise_bf = (__bf16*)alloc((size_t)BATCH * 64 * 2);
  float* beffF = (float*)alloc((size_t)TSTEP * MLPW * 4);
  float* beffG = (float*)alloc((size_t)TSTEP * MLPW * 4);
  float* ybuf[2]  = {(float*)alloc((size_t)BATCH * HIDD * 4), (float*)alloc((size_t)BATCH * HIDD * 4)};
  float* yhbuf[2] = {(float*)alloc((size_t)BATCH * HIDD * 4), (float*)alloc((size_t)BATCH * HIDD * 4)};
  float* fbuf[2]  = {(float*)alloc((size_t)BATCH * HIDD * 4), (float*)alloc((size_t)BATCH * HIDD * 4)};
  float* gdwb  = (float*)alloc((size_t)BATCH * HIDD * 4);
  float* gbuf  = (float*)alloc((size_t)BATCH * HIDD * NOISE * 4);
  float* xs    = (float*)alloc((size_t)BATCH * TSTEP * HIDD * 4);

  // ---- one-time weight conversion ----
  auto cwt = [&](const float* W, __bf16* Wt, int K, int N, int rowOff) {
    int n = K * N;
    conv_wt_kernel<<<(n + 255) / 256, 256, 0, stream>>>(W, Wt, K, N, rowOff);
  };
  cwt(iW0, iW0t, 64, 512, 0);
  cwt(iWh,             iWh0t, 512, 512, 0);
  cwt(iWh + 512 * 512, iWh1t, 512, 512, 0);
  cwt(iWo, iWot, 512, 128, 0);
  cwt(fW0, fW0t, 128, 512, 1);            // skip t-row (folded into bias)
  cwt(fWh,             fWh0t, 512, 512, 0);
  cwt(fWh + 512 * 512, fWh1t, 512, 512, 0);
  cwt(fWo, fWot, 512, 128, 0);
  cwt(gW0, gW0t, 128, 512, 1);
  cwt(gWh,             gWh0t, 512, 512, 0);
  cwt(gWh + 512 * 512, gWh1t, 512, 512, 0);
  cwt(gWo, gWot, 512, 2048, 0);
  conv_bf_kernel<<<(BATCH * 64 + 255) / 256, 256, 0, stream>>>(noise, noise_bf, BATCH * 64);
  bias_eff_kernel<<<(TSTEP * MLPW + 255) / 256, 256, 0, stream>>>(fb0, fW0, ts, beffF);
  bias_eff_kernel<<<(TSTEP * MLPW + 255) / 256, 256, 0, stream>>>(gb0, gW0, ts, beffG);

  // ---- GEMM launcher ----
  auto gemm = [&](const __bf16* Ain, const __bf16* Wt, const float* bias, int K, int N,
                  int act, float* oF, __bf16* oB) {
    dim3 grid(N / 128, BATCH / 64);
    if (act == 1)
      gemm_wmma_kernel<1, false, true><<<grid, 256, 0, stream>>>(Ain, Wt, bias, nullptr, oB, K, N);
    else if (act == 0)
      gemm_wmma_kernel<0, true, true><<<grid, 256, 0, stream>>>(Ain, Wt, bias, oF, oB, K, N);
    else
      gemm_wmma_kernel<2, true, false><<<grid, 256, 0, stream>>>(Ain, Wt, bias, oF, nullptr, K, N);
  };
  auto mlp = [&](const __bf16* xin, int Kin,
                 const __bf16* W0t, const float* b0,
                 const __bf16* Wh0t, const float* bh0,
                 const __bf16* Wh1t, const float* bh1,
                 const __bf16* Wot, const float* bo, int Nout,
                 int outAct, float* oF, __bf16* oB) {
    gemm(xin,  W0t,  b0,  Kin, 512,  1, nullptr, actA);
    gemm(actA, Wh0t, bh0, 512, 512,  1, nullptr, actB);
    gemm(actB, Wh1t, bh1, 512, 512,  1, nullptr, actA);
    gemm(actA, Wot,  bo,  512, Nout, outAct, oF, oB);
  };

  const int EW_BLOCKS = (BATCH * HIDD + 255) / 256;

  // ---- x0 = MLP_i(init_noise); y0 = yhat0 = x0 ----
  mlp(noise_bf, 64, iW0t, ib0, iWh0t, ibh, iWh1t, ibh + 512, iWot, ibo, 128,
      /*none*/0, ybuf[0], x_bf);
  init_state_kernel<<<EW_BLOCKS, 256, 0, stream>>>(ybuf[0], yhbuf[0], xs);

  // ---- f0, g0 at ts[0] ----
  mlp(x_bf, 128, fW0t, beffF + 0 * MLPW, fWh0t, fbh, fWh1t, fbh + 512, fWot, fbo, 128,
      /*tanh*/2, fbuf[0], nullptr);
  mlp(x_bf, 128, gW0t, beffG + 0 * MLPW, gWh0t, gbh, gWh1t, gbh + 512, gWot, gbo, 2048,
      /*tanh*/2, gbuf, nullptr);

  // ---- sequential scan over 63 steps ----
  for (int k = 0; k < TSTEP - 1; ++k) {
    const int cur = k & 1, nxt = (k + 1) & 1;
    const float* dWk = dW + (size_t)k * BATCH * NOISE;

    step_yhat_kernel<<<EW_BLOCKS, 256, 0, stream>>>(
        ybuf[cur], yhbuf[cur], fbuf[cur], gbuf, dWk, ts, k,
        gdwb, yhbuf[nxt], x_bf);

    mlp(x_bf, 128, fW0t, beffF + (size_t)(k + 1) * MLPW, fWh0t, fbh, fWh1t, fbh + 512,
        fWot, fbo, 128, 2, fbuf[nxt], nullptr);
    mlp(x_bf, 128, gW0t, beffG + (size_t)(k + 1) * MLPW, gWh0t, gbh, gWh1t, gbh + 512,
        gWot, gbo, 2048, 2, gbuf, nullptr);

    step_y_kernel<<<EW_BLOCKS, 256, 0, stream>>>(
        ybuf[cur], fbuf[cur], fbuf[nxt], gdwb, gbuf, dWk, ts, k,
        ybuf[nxt], xs);
  }

  // ---- readout ----
  readout_kernel<<<(BATCH * TSTEP + 255) / 256, 256, 0, stream>>>(xs, ts, rW, rb, out);
}
